// ConvLayer_53541062312240
// MI455X (gfx1250) — matrix operand order repaired and checked by reading, not compiled, not run on hardware
//
#include <hip/hip_runtime.h>
#include <hip/hip_bf16.h>
#include <math.h>

typedef __attribute__((ext_vector_type(16))) __bf16 v16bf;
typedef __attribute__((ext_vector_type(8)))  float  v8f;
typedef __attribute__((ext_vector_type(4)))  unsigned int u32x4;
typedef __attribute__((ext_vector_type(8)))  int i32x8;
typedef __attribute__((ext_vector_type(4)))  int i32x4;

#define N_NODES 50000
#define N_EDGES 1200000
#define DIM     64
#define FAN_IN  192
#define BN_EPS  1e-5f

#define MSG_FLOATS   (N_NODES * DIM)
#define STAT_FLOATS  128
#define WPACK_HALFS  (2 * 4 * 6 * 32 * 16)   // mats * ntiles * kchunks * lanes * 16
#define N_TILES      (N_EDGES / 16)          // 75000
#define EDGE_BLOCKS  2048

static __device__ __forceinline__ unsigned short f32_to_bf16(float f) {
    unsigned int u = __float_as_uint(f);
    u += 0x7fffu + ((u >> 16) & 1u);     // round to nearest even
    return (unsigned short)(u >> 16);
}

static __device__ __forceinline__ float softplus_f(float v) {
    return fmaxf(v, 0.f) + __logf(1.f + __expf(-fabsf(v)));
}

static __device__ __forceinline__ void atomAddF(float* p, float v) {
#if defined(__HIP_DEVICE_COMPILE__)
    unsafeAtomicAdd(p, v);               // hardware global_atomic_add_f32
#else
    atomicAdd(p, v);
#endif
}

// ---------------- kernel 0: zero message + stats ----------------
__global__ void k_init(float* ws) {
    int i = blockIdx.x * blockDim.x + threadIdx.x;
    if (i < MSG_FLOATS + STAT_FLOATS) ws[i] = 0.f;
}

// ---------------- kernel 1: pack Wf/Ws into bf16 WMMA B-fragment layout ----
__global__ void k_pack(const float* __restrict__ Wf, const float* __restrict__ Ws,
                       unsigned short* __restrict__ wpack) {
    int t = blockIdx.x * blockDim.x + threadIdx.x;      // 48 frags * 32 lanes
    if (t >= 48 * 32) return;
    int frag = t >> 5;
    int lane = t & 31;
    int mat   = frag / 24;
    int rem   = frag % 24;
    int ntile = rem / 6;
    int kc    = rem % 6;
    const float* W = mat ? Ws : Wf;
    int n = ntile * 16 + (lane & 15);
    unsigned short* dst = wpack + frag * 512 + lane * 16;
#pragma unroll
    for (int j = 0; j < 16; ++j) {
        int k = kc * 32 + (j < 8 ? (lane >> 4) * 8 + j : 16 + (lane >> 4) * 8 + (j - 8));
        dst[j] = f32_to_bf16(W[n * FAN_IN + k]);
    }
}

// ---------------- kernel 2: persistent edge GEMMs (bf16 WMMA) + scatter ----
// Weights end up register-resident (48 frags * 8 VGPRs, CDNA5 1024-VGPR file);
// __launch_bounds__(256,1) gives the allocator the full budget (no spills).
__global__ __launch_bounds__(256, 1) void k_edge(
    const float* __restrict__ x, const float* __restrict__ edge_attr,
    const float* __restrict__ bfv, const float* __restrict__ bsv,
    const int* __restrict__ esrc, const int* __restrict__ etgt,
    const unsigned short* __restrict__ wpack, float* __restrict__ message) {

    __shared__ unsigned short ldsW[WPACK_HALFS];        // 49,152 B

    const int lane  = threadIdx.x & 31;
    const int wave  = threadIdx.x >> 5;
    const int khalf = lane >> 4;                        // A-layout half-group
    const int row   = lane & 15;

    // ---- stage packed weights into LDS via the Tensor Data Mover ----
#if __has_builtin(__builtin_amdgcn_tensor_load_to_lds)
    if (wave == 0) {
        unsigned int ldsAddr = (unsigned int)(unsigned long long)&ldsW[0];
        unsigned long long ga = (unsigned long long)wpack;
        u32x4 g0;
        g0[0] = 1u;                                     // count=1, user mode
        g0[1] = ldsAddr;                                // lds_addr [63:32]
        g0[2] = (unsigned int)(ga & 0xffffffffull);     // global_addr lo
        g0[3] = (unsigned int)((ga >> 32) & 0x01ffffffu) | (2u << 30); // addr hi | type=2
        i32x8 g1;
        g1[0] = 3 << 16;        // wg_mask=0, data_size=3 (8B), no flags
        g1[1] = 0x1800 << 16;   // tensor_dim0[15:0]=6144 in bits 63:48
        g1[2] = 1 << 16;        // tensor_dim0 hi=0, tensor_dim1=1
        g1[3] = 0x1800 << 16;   // tensor_dim1 hi=0, tile_dim0=6144
        g1[4] = 1;              // tile_dim1=1, tile_dim2=0
        g1[5] = 6144;           // tensor_dim0_stride lo
        g1[6] = 0;              // stride hi, dim1_stride lo
        g1[7] = 0;
        i32x4 gz4 = {0, 0, 0, 0};
        i32x8 gz8 = {0, 0, 0, 0, 0, 0, 0, 0};
        __builtin_amdgcn_tensor_load_to_lds(g0, g1, gz4, gz4, gz8, 0);
        __builtin_amdgcn_s_wait_tensorcnt(0);
    }
#else
    {
        const uint4* g = (const uint4*)wpack;
        uint4* l = (uint4*)ldsW;
        for (int i = threadIdx.x; i < (WPACK_HALFS / 8); i += 256) l[i] = g[i];
    }
#endif
    __syncthreads();

    // hoist biases: n = t*16 + row, invariant across tiles
    float bF[4], bS[4];
#pragma unroll
    for (int t = 0; t < 4; ++t) { bF[t] = bfv[t * 16 + row]; bS[t] = bsv[t * 16 + row]; }

    const int tileStride = EDGE_BLOCKS * 8;
    for (int tile = blockIdx.x * 8 + wave; tile < N_TILES; tile += tileStride) {
        const int tileBase = tile * 16;
        const int e = tileBase + row;

        const int src = esrc[e];
        const int tgt = etgt[e];

        // prefetch next tile's edge_attr stream (speculative, HBM-resident)
        {
            int ne = e + tileStride * 16;
            if (ne < N_EDGES)
                __builtin_prefetch(edge_attr + (size_t)ne * DIM, 0, 0);
        }

        const float* px_s = x + (size_t)src * DIM;
        const float* px_t = x + (size_t)tgt * DIM;
        const float* pea  = edge_attr + (size_t)e * DIM;

        v8f accF[4] = {v8f{}, v8f{}, v8f{}, v8f{}};
        v8f accS[4] = {v8f{}, v8f{}, v8f{}, v8f{}};

#pragma unroll
        for (int c = 0; c < 6; ++c) {
            const float* base = (c < 2) ? px_s : (c < 4) ? px_t : pea;   // compile-time c
            const float* p = base + (c & 1) * 32 + khalf * 8;
            float4 fa = ((const float4*)p)[0];
            float4 fb = ((const float4*)p)[1];
            float4 fc = ((const float4*)(p + 16))[0];
            float4 fd = ((const float4*)(p + 16))[1];
            v16bf a;
            a[0]  = (__bf16)fa.x; a[1]  = (__bf16)fa.y; a[2]  = (__bf16)fa.z; a[3]  = (__bf16)fa.w;
            a[4]  = (__bf16)fb.x; a[5]  = (__bf16)fb.y; a[6]  = (__bf16)fb.z; a[7]  = (__bf16)fb.w;
            a[8]  = (__bf16)fc.x; a[9]  = (__bf16)fc.y; a[10] = (__bf16)fc.z; a[11] = (__bf16)fc.w;
            a[12] = (__bf16)fd.x; a[13] = (__bf16)fd.y; a[14] = (__bf16)fd.z; a[15] = (__bf16)fd.w;

#pragma unroll
            for (int mat = 0; mat < 2; ++mat) {
#pragma unroll
                for (int t = 0; t < 4; ++t) {
                    int frag = (mat * 4 + t) * 6 + c;
                    const uint4* bp = (const uint4*)(ldsW + frag * 512 + lane * 16);
                    union { uint4 u[2]; v16bf v; } bw;
                    bw.u[0] = bp[0];
                    bw.u[1] = bp[1];
                    v8f& acc = mat ? accS[t] : accF[t];
                    acc = __builtin_amdgcn_wmma_f32_16x16x32_bf16(
                        false, a, false, bw.v, (short)0, acc, false, false);
                }
            }
        }

        // epilogue: gate=sigmoid, val=softplus, scatter-add to message[src]
        // C/D layout: lane -> N = t*16 + row ; VGPR r -> M = khalf*8 + r
#pragma unroll
        for (int t = 0; t < 4; ++t) {
            int n = t * 16 + row;
#pragma unroll
            for (int r = 0; r < 8; ++r) {
                int m = khalf * 8 + r;
                float g = accF[t][r] + bF[t];
                g = 1.f / (1.f + __expf(-g));
                float v = softplus_f(accS[t][r] + bS[t]);
                int sidx = esrc[tileBase + m];
                atomAddF(message + (size_t)sidx * DIM + n, g * v);
            }
        }
    }
}

// ---------------- kernel 3: BN statistics (sum / sumsq per column) ---------
__global__ void k_stats(const float* __restrict__ msg, float* __restrict__ stats) {
    int t = blockIdx.x * blockDim.x + threadIdx.x;
    int col = t & 63;
    int row = t >> 6;
    int rstride = (gridDim.x * blockDim.x) >> 6;
    float s = 0.f, q = 0.f;
    for (int r = row; r < N_NODES; r += rstride) {
        float v = msg[(size_t)r * DIM + col];
        s += v; q += v * v;
    }
    atomAddF(stats + col, s);
    atomAddF(stats + 64 + col, q);
}

// ---------------- kernel 4: normalize + softplus residual ------------------
__global__ void k_final(const float* __restrict__ x, const float* __restrict__ msg,
                        const float* __restrict__ stats,
                        const float* __restrict__ gamma, const float* __restrict__ beta,
                        float* __restrict__ out) {
    int i = blockIdx.x * blockDim.x + threadIdx.x;
    if (i >= MSG_FLOATS) return;
    int col = i & 63;
    const float invN = 1.f / (float)N_NODES;
    float mean = stats[col] * invN;
    float var  = stats[64 + col] * invN - mean * mean;
    float inv  = rsqrtf(var + BN_EPS);
    float m = (msg[i] - mean) * inv * gamma[col] + beta[col];
    out[i] = softplus_f(x[i] + m);
}

extern "C" void kernel_launch(void* const* d_in, const int* in_sizes, int n_in,
                              void* d_out, int out_size, void* d_ws, size_t ws_size,
                              hipStream_t stream) {
    const float* x         = (const float*)d_in[0];
    const float* edge_attr = (const float*)d_in[1];
    const float* Wf        = (const float*)d_in[2];
    const float* bf        = (const float*)d_in[3];
    const float* Ws        = (const float*)d_in[4];
    const float* bs        = (const float*)d_in[5];
    const float* gamma     = (const float*)d_in[6];
    const float* beta      = (const float*)d_in[7];
    const int*   esrc      = (const int*)d_in[8];
    const int*   etgt      = (const int*)d_in[9];
    float* out = (float*)d_out;

    float* msg   = (float*)d_ws;
    float* stats = msg + MSG_FLOATS;
    unsigned short* wpack = (unsigned short*)(stats + STAT_FLOATS);

    k_init<<<(MSG_FLOATS + STAT_FLOATS + 255) / 256, 256, 0, stream>>>(msg);
    k_pack<<<(48 * 32 + 255) / 256, 256, 0, stream>>>(Wf, Ws, wpack);
    k_edge<<<EDGE_BLOCKS, 256, 0, stream>>>(x, edge_attr, bf, bs, esrc, etgt, wpack, msg);
    k_stats<<<256, 256, 0, stream>>>(msg, stats);
    k_final<<<(MSG_FLOATS + 255) / 256, 256, 0, stream>>>(x, msg, stats, gamma, beta, out);
}
